// DepthwiseXCorr_49589692399972
// MI455X (gfx1250) — matrix operand order repaired and checked by reading, not compiled, not run on hardware
//
#include <hip/hip_runtime.h>

// ---------------------------------------------------------------------------
// DepthwiseXCorr head for MI455X (gfx1250, wave32, WMMA + TDM).
// Pipeline: prep (BN fold) -> kbranch (dw + WMMA GEMM) -> sdw -> sgemm (WMMA)
//           -> xcorr. 1x1 convs run on v_wmma_f32_16x16x32_f16 (f32 accum).
// Staging copies in sdw/xcorr use the Tensor Data Mover when available.
// With 192MB L2, all intermediates (~113MB) stay cache-resident; HBM traffic
// ~= inputs (132MB) + output (82MB) -> near the 23.3TB/s roofline floor.
// ---------------------------------------------------------------------------

typedef __attribute__((ext_vector_type(16))) _Float16 v16h;
typedef __attribute__((ext_vector_type(8)))  _Float16 v8h;
typedef __attribute__((ext_vector_type(8)))  float    v8f;
typedef __attribute__((ext_vector_type(4)))  unsigned int v4u;
typedef __attribute__((ext_vector_type(8)))  int      v8i;
typedef __attribute__((ext_vector_type(4)))  int      v4i;

#define NB   128   // batch
#define NC   256   // channels
#define KH   7
#define SH   31
#define KOH  5     // 5x5  kernel-branch spatial
#define SOH  29    // 29x29 search-branch spatial
#define SP   841   // 29*29
#define SPAD 848   // 841 padded to 16*53
#define NMT  53    // m-tiles of 16 over 848
#define OO   25    // 25x25 output spatial
#define OP   625
#define LDSA 264   // LDS A-tile row pitch (halves): conflict-free, 16B aligned

#if defined(__gfx1250__) && __has_builtin(__builtin_amdgcn_tensor_load_to_lds) && \
    __has_builtin(__builtin_amdgcn_s_wait_tensorcnt)
#define HAS_TDM 1
#else
#define HAS_TDM 0
#endif

#if HAS_TDM
// 1-D TDM copy: ndwords dwords from global -> LDS, wave-uniform addresses.
// D# per CDNA5 ISA ch.8: group0 = {count=1, lds_addr, global_addr, type=2},
// group1 = {data_size=4B, tensor_dim0=tile_dim0=ndwords, tensor_dim1=1}.
__device__ __forceinline__ void tdm_load_1d(const void* gptr, void* lptr,
                                            unsigned ndwords) {
    unsigned long long ga = (unsigned long long)(size_t)gptr;
    unsigned la = (unsigned)(size_t)lptr;  // LDS aperture: [31:0] = LDS offset
    v4u g0;
    g0[0] = 1u;                                  // count=1 (valid user D#)
    g0[1] = la;                                  // lds_addr
    g0[2] = (unsigned)ga;                        // global_addr[31:0]
    g0[3] = (unsigned)((ga >> 32) & 0x1FFFFFFu)  // global_addr[56:32]
            | (2u << 30);                        // type=2 ("image")
    v8i g1;
    g1[0] = (int)(2u << 16);                        // data_size=2 -> 4 bytes
    g1[1] = (int)((ndwords & 0xFFFFu) << 16);       // tensor_dim0[15:0]
    g1[2] = (int)(((ndwords >> 16) & 0xFFFFu)       // tensor_dim0[31:16]
            | (1u << 16));                          // tensor_dim1[15:0]=1
    g1[3] = (int)((ndwords & 0xFFFFu) << 16);       // tile_dim0 = ndwords
    g1[4] = 0;                                      // tile_dim1=0, tile_dim2=0
    g1[5] = (int)ndwords;                           // tensor_dim0_stride lo
    g1[6] = 0;
    g1[7] = 0;
    v4i z4 = {0, 0, 0, 0};
#if __clang_major__ >= 23
    v8i z8 = {0, 0, 0, 0, 0, 0, 0, 0};
    __builtin_amdgcn_tensor_load_to_lds(g0, g1, z4, z4, z8, 0);
#else
    __builtin_amdgcn_tensor_load_to_lds(g0, g1, z4, z4, 0);
#endif
}
#endif

// ---------------------------------------------------------------------------
// Kernel 1: fold BN params. grid 512 (256 per branch), block 256.
// ---------------------------------------------------------------------------
__global__ __launch_bounds__(256) void dx_prep(
    const float* __restrict__ k_dw_w, const float* __restrict__ k_bn1,
    const float* __restrict__ k_pw_w, const float* __restrict__ k_pw_b,
    const float* __restrict__ k_bn2,
    const float* __restrict__ s_dw_w, const float* __restrict__ s_bn1,
    const float* __restrict__ s_pw_w, const float* __restrict__ s_pw_b,
    const float* __restrict__ s_bn2,
    _Float16* __restrict__ Wk16, _Float16* __restrict__ Ws16,
    float* __restrict__ biasK, float* __restrict__ biasS,
    float* __restrict__ dwKf, float* __restrict__ dwSf)
{
    const int  o   = blockIdx.x & 255;
    const bool sbr = blockIdx.x >= 256;
    const float* bn1 = sbr ? s_bn1 : k_bn1;
    const float* bn2 = sbr ? s_bn2 : k_bn2;
    const float* pw  = sbr ? s_pw_w : k_pw_w;
    const float* pwb = sbr ? s_pw_b : k_pw_b;
    const float* dw  = sbr ? s_dw_w : k_dw_w;
    _Float16* W16  = sbr ? Ws16 : Wk16;
    float*    bias = sbr ? biasS : biasK;
    float*    dwf  = sbr ? dwSf : dwKf;

    const int i = threadIdx.x;
    const float inv2 = bn2[o] / sqrtf(bn2[3 * NC + o] + 1e-5f);
    W16[o * NC + i] = (_Float16)(pw[o * NC + i] * inv2);
    if (i == 0)
        bias[o] = pwb[o] * inv2 + bn2[NC + o] - bn2[2 * NC + o] * inv2;

    const float inv1 = bn1[o] / sqrtf(bn1[3 * NC + o] + 1e-5f);
    if (i < 9) dwf[o * 10 + i] = dw[o * 9 + i] * inv1;
    if (i == 9) dwf[o * 10 + 9] = bn1[NC + o] - bn1[2 * NC + o] * inv1;
}

// Load a 16x32 f16 A-fragment from an LDS tile row block (contiguous k order, k = 16*h + s).
__device__ __forceinline__ v16h load_a_frag(const _Float16* base) {
    v8h lo = *(const v8h*)(base);
    v8h hi = *(const v8h*)(base + 8);
    v16h a;
#pragma unroll
    for (int j = 0; j < 8; ++j) { a[j] = lo[j]; a[j + 8] = hi[j]; }
    return a;
}

// ---------------------------------------------------------------------------
// Kernel 2: kernel branch, fused dw(3x3)+BN+ReLU6 -> WMMA GEMM -> f32 maps.
// One block per batch; M=25 (pad 32), N=256, K=256.
// ---------------------------------------------------------------------------
__global__ __launch_bounds__(256) void dx_kbranch(
    const float* __restrict__ xin, const _Float16* __restrict__ W16,
    const float* __restrict__ bias, const float* __restrict__ dwf,
    float* __restrict__ kOut)
{
    __shared__ _Float16 As[32][LDSA];
    const int b = blockIdx.x;
    const int t = threadIdx.x;

    for (int idx = t; idx < 32 * LDSA; idx += 256)
        (&As[0][0])[idx] = (_Float16)0.0f;
    __syncthreads();

    {   // thread t handles channel c = t : 7x7 -> 5x5 dw conv + BN + ReLU6
        const int c = t;
        const float* xp = xin + ((size_t)b * NC + c) * (KH * KH);
        float xr[KH * KH];
#pragma unroll
        for (int j = 0; j < KH * KH; ++j) xr[j] = xp[j];
        float w[9];
#pragma unroll
        for (int j = 0; j < 9; ++j) w[j] = dwf[c * 10 + j];
        const float sh = dwf[c * 10 + 9];
#pragma unroll
        for (int oy = 0; oy < KOH; ++oy)
#pragma unroll
            for (int ox = 0; ox < KOH; ++ox) {
                float acc = sh;
#pragma unroll
                for (int ky = 0; ky < 3; ++ky)
#pragma unroll
                    for (int kx = 0; kx < 3; ++kx)
                        acc += w[ky * 3 + kx] * xr[(oy + ky) * KH + ox + kx];
                acc = fminf(fmaxf(acc, 0.0f), 6.0f);
                As[oy * KOH + ox][c] = (_Float16)acc;
            }
    }
    __syncthreads();

    const int wave = t >> 5, lane = t & 31;
    const int ln = lane & 15, lh = lane >> 4;
    const int nt0 = wave * 2;

    v8f acc[2][2];
#pragma unroll
    for (int mt = 0; mt < 2; ++mt)
#pragma unroll
        for (int nl = 0; nl < 2; ++nl)
#pragma unroll
            for (int r = 0; r < 8; ++r) acc[mt][nl][r] = 0.0f;

    for (int kc = 0; kc < 8; ++kc) {
        v16h a[2], bm[2];
#pragma unroll
        for (int mt = 0; mt < 2; ++mt)
            a[mt] = load_a_frag(&As[mt * 16 + ln][kc * 32 + lh * 16]);
#pragma unroll
        for (int nl = 0; nl < 2; ++nl) {
            const int n = (nt0 + nl) * 16 + ln;
            bm[nl] = *(const v16h*)(W16 + (size_t)n * NC + kc * 32 + lh * 16);
        }
#pragma unroll
        for (int mt = 0; mt < 2; ++mt)
#pragma unroll
            for (int nl = 0; nl < 2; ++nl)
                acc[mt][nl] = __builtin_amdgcn_wmma_f32_16x16x32_f16(
                    false, a[mt], false, bm[nl], (short)0, acc[mt][nl],
                    false, false);
    }

#pragma unroll
    for (int mt = 0; mt < 2; ++mt)
#pragma unroll
        for (int nl = 0; nl < 2; ++nl) {
            const int n = (nt0 + nl) * 16 + ln;
            const float bs = bias[n];
#pragma unroll
            for (int r = 0; r < 8; ++r) {
                const int m = mt * 16 + lh * 8 + r;
                if (m < KOH * KOH)
                    kOut[((size_t)b * NC + n) * (KOH * KOH) + m] =
                        acc[mt][nl][r] + bs;
            }
        }
}

// ---------------------------------------------------------------------------
// Kernel 3: search branch dw(3x3)+BN+ReLU6 -> f16 activations [b][c][848].
// One block per (b,c) plane. Input plane staged to LDS via TDM when available.
// ---------------------------------------------------------------------------
__global__ __launch_bounds__(256) void dx_sdw(
    const float* __restrict__ xin, const float* __restrict__ dwf,
    _Float16* __restrict__ actS)
{
    __shared__ float xs[SH * SH];
    const int bc = blockIdx.x;
    const int c  = bc & 255;
    const int t  = threadIdx.x;

    const float* xp = xin + (size_t)bc * (SH * SH);
#if HAS_TDM
    if (t == 0) {  // one TDM issue (wave0); TENSORcnt drained before barrier
        tdm_load_1d(xp, xs, SH * SH);
        __builtin_amdgcn_s_wait_tensorcnt(0);
    }
    __syncthreads();
#else
    for (int idx = t; idx < SH * SH; idx += 256) xs[idx] = xp[idx];
    __syncthreads();
#endif

    float w[9];
#pragma unroll
    for (int j = 0; j < 9; ++j) w[j] = dwf[c * 10 + j];
    const float sh = dwf[c * 10 + 9];

    _Float16* op = actS + (size_t)bc * SPAD;
    for (int p = t; p < SP; p += 256) {
        const int oy = p / SOH, ox = p % SOH;
        const float* row = &xs[oy * SH + ox];
        float acc = sh;
#pragma unroll
        for (int ky = 0; ky < 3; ++ky)
#pragma unroll
            for (int kx = 0; kx < 3; ++kx)
                acc += w[ky * 3 + kx] * row[ky * SH + kx];
        acc = fminf(fmaxf(acc, 0.0f), 6.0f);
        op[p] = (_Float16)acc;
    }
}

// ---------------------------------------------------------------------------
// Kernel 4: search 1x1 conv as WMMA GEMM. Block = (batch, 16-row p-tile).
// 8 waves x 2 N-tiles cover N=256; K-loop 8 x 32. s stored as f16 [b][c][848].
// ---------------------------------------------------------------------------
__global__ __launch_bounds__(256) void dx_sgemm(
    const _Float16* __restrict__ actS, const _Float16* __restrict__ W16,
    const float* __restrict__ bias, _Float16* __restrict__ s16)
{
    __shared__ _Float16 As[16][LDSA];
    const int blk = blockIdx.x;
    const int b  = blk / NMT;
    const int mt = blk - b * NMT;
    const int p0 = mt * 16;
    const int t  = threadIdx.x;

    {   // stage A^T: thread t = channel c copies 16 contiguous halves
        const int c = t;
        const v16h v = *(const v16h*)(actS + ((size_t)b * NC + c) * SPAD + p0);
#pragma unroll
        for (int p = 0; p < 16; ++p) As[p][c] = v[p];
    }
    __syncthreads();

    const int wave = t >> 5, lane = t & 31;
    const int ln = lane & 15, lh = lane >> 4;
    const int nt0 = wave * 2;

    v8f acc[2];
#pragma unroll
    for (int nl = 0; nl < 2; ++nl)
#pragma unroll
        for (int r = 0; r < 8; ++r) acc[nl][r] = 0.0f;

    for (int kc = 0; kc < 8; ++kc) {
        const v16h a = load_a_frag(&As[ln][kc * 32 + lh * 16]);
#pragma unroll
        for (int nl = 0; nl < 2; ++nl) {
            const int n = (nt0 + nl) * 16 + ln;
            if (kc < 7)  // uniform branch: prefetch next weight panel
                __builtin_prefetch(W16 + (size_t)n * NC + (kc + 1) * 32 + lh * 16, 0, 0);
            const v16h bm =
                *(const v16h*)(W16 + (size_t)n * NC + kc * 32 + lh * 16);
            acc[nl] = __builtin_amdgcn_wmma_f32_16x16x32_f16(
                false, a, false, bm, (short)0, acc[nl], false, false);
        }
    }

#pragma unroll
    for (int nl = 0; nl < 2; ++nl) {
        const int n = (nt0 + nl) * 16 + ln;
        const float bs = bias[n];
        v8h outv;
#pragma unroll
        for (int r = 0; r < 8; ++r) outv[r] = (_Float16)(acc[nl][r] + bs);
        // rows m = lh*8 + r are contiguous p values -> one 16B store
        *(v8h*)(s16 + ((size_t)b * NC + n) * SPAD + p0 + lh * 8) = outv;
    }
}

// ---------------------------------------------------------------------------
// Kernel 5: per-(b,c) depthwise cross-correlation 29x29 (*) 5x5 -> 25x25 f32.
// s16 plane staged LDS-raw via TDM (1696B tile), then converted to f32.
// ---------------------------------------------------------------------------
__global__ __launch_bounds__(256) void dx_xcorr(
    const _Float16* __restrict__ s16, const float* __restrict__ kOut,
    float* __restrict__ out)
{
    __shared__ float sp[SP + 7];
    __shared__ float kf[KOH * KOH];
    const int bc = blockIdx.x;
    const int t  = threadIdx.x;

    const _Float16* spg = s16 + (size_t)bc * SPAD;
#if HAS_TDM
    __shared__ _Float16 sp16[SPAD];
    if (t == 0) {  // DMA the raw f16 plane (424 dwords) into LDS
        tdm_load_1d(spg, sp16, SPAD / 2);
        __builtin_amdgcn_s_wait_tensorcnt(0);
    }
    if (t < KOH * KOH) kf[t] = kOut[(size_t)bc * (KOH * KOH) + t];
    __syncthreads();
    for (int idx = t; idx < SP; idx += 256) sp[idx] = (float)sp16[idx];
    __syncthreads();
#else
    for (int idx = t; idx < SP; idx += 256) sp[idx] = (float)spg[idx];
    if (t < KOH * KOH) kf[t] = kOut[(size_t)bc * (KOH * KOH) + t];
    __syncthreads();
#endif

    float* op = out + (size_t)bc * OP;
    for (int p = t; p < OP; p += 256) {
        const int oy = p / OO, ox = p % OO;
        float acc = 0.0f;
#pragma unroll
        for (int ky = 0; ky < KOH; ++ky) {
            const float* row = &sp[(oy + ky) * SOH + ox];
#pragma unroll
            for (int kx = 0; kx < KOH; ++kx)
                acc += kf[ky * KOH + kx] * row[kx];
        }
        op[p] = acc;
    }
}

// ---------------------------------------------------------------------------
// Host launch
// ---------------------------------------------------------------------------
extern "C" void kernel_launch(void* const* d_in, const int* in_sizes, int n_in,
                              void* d_out, int out_size, void* d_ws, size_t ws_size,
                              hipStream_t stream) {
    (void)in_sizes; (void)n_in; (void)out_size; (void)ws_size;

    const float* kernel_in = (const float*)d_in[0];
    const float* search_in = (const float*)d_in[1];
    const float* k_dw_w = (const float*)d_in[2];
    const float* k_bn1  = (const float*)d_in[3];
    const float* k_pw_w = (const float*)d_in[4];
    const float* k_pw_b = (const float*)d_in[5];
    const float* k_bn2  = (const float*)d_in[6];
    const float* s_dw_w = (const float*)d_in[7];
    const float* s_bn1  = (const float*)d_in[8];
    const float* s_pw_w = (const float*)d_in[9];
    const float* s_pw_b = (const float*)d_in[10];
    const float* s_bn2  = (const float*)d_in[11];
    float* out = (float*)d_out;

    // workspace layout (bytes, 256-aligned)
    char* ws = (char*)d_ws;
    _Float16* Wk16  = (_Float16*)(ws);                       //   131072
    _Float16* Ws16  = (_Float16*)(ws + 131072);              //   131072
    float*    biasK = (float*)(ws + 262144);                 //     1024
    float*    biasS = (float*)(ws + 263168);                 //     1024
    float*    dwKf  = (float*)(ws + 264192);                 //    10240
    float*    dwSf  = (float*)(ws + 274432);                 //    10240
    float*    kOut  = (float*)(ws + 284672);                 //  3276800
    _Float16* actS  = (_Float16*)(ws + 3561472);             // 55574528
    _Float16* s16   = (_Float16*)(ws + 59136000);            // 55574528
    // total: 114,710,528 bytes

    dx_prep<<<dim3(512), dim3(256), 0, stream>>>(
        k_dw_w, k_bn1, k_pw_w, k_pw_b, k_bn2,
        s_dw_w, s_bn1, s_pw_w, s_pw_b, s_bn2,
        Wk16, Ws16, biasK, biasS, dwKf, dwSf);

    dx_kbranch<<<dim3(NB), dim3(256), 0, stream>>>(
        kernel_in, Wk16, biasK, dwKf, kOut);

    dx_sdw<<<dim3(NB * NC), dim3(256), 0, stream>>>(
        search_in, dwSf, actS);

    dx_sgemm<<<dim3(NB * NMT), dim3(256), 0, stream>>>(
        actS, Ws16, biasS, s16);

    dx_xcorr<<<dim3(NB * NC), dim3(256), 0, stream>>>(
        s16, kOut, out);
}
